// ScaledDotProductAttention_3582002725231
// MI455X (gfx1250) — matrix operand order repaired and checked by reading, not compiled, not run on hardware
//
#include <hip/hip_runtime.h>
#include <hip/hip_bf16.h>

typedef __attribute__((ext_vector_type(16))) _Float16 v16h;
typedef __attribute__((ext_vector_type(8)))  float    v8f;

#define BH_N   16
#define LSEQ   2048
#define DDIM   64
#define SCALE  0.125f            // 1 / temperature (temperature = 8.0)
#define MFILL  -65504.0f

// LDS strides, padded for alignment + bank spread
#define KSTR4   72   // K tile (f32): 32 keys x 64 d, floats/row (288B, 16B-aligned chunks)
#define VSTRIDE 48   // V tile transposed (f16): 64 d x 32 keys (96B/row, 32B-aligned chunks)
#define PSTRIDE 48   // P tile per wave (f16): 16 rows x 32 cols
#define MSTR    32   // mask tile: 64 q-rows x 32 key-cols (bytes)

// ---- async global->LDS staging of K (f32, no VGPR round trip, ASYNCcnt) ----
__device__ __forceinline__ void stage_K_async(const float* __restrict__ Kb, int k0,
                                              float* __restrict__ kbuf, int tid) {
    #pragma unroll
    for (int i = 0; i < 4; ++i) {
        int idx4 = (i * 128 + tid) * 4;            // 0..2044 step 4
        int key = idx4 >> 6, d = idx4 & 63;
        const float* g = Kb + (size_t)(k0 + key) * DDIM + d;
        // low 32 bits of a generic LDS pointer == LDS byte address
        unsigned ldsa = (unsigned)(unsigned long long)(kbuf + key * KSTR4 + d);
        asm volatile("global_load_async_to_lds_b128 %0, %1, off"
                     :: "v"(ldsa), "v"(g) : "memory");
    }
}

// ---- async staging of the 64x32 mask tile for this block ----
__device__ __forceinline__ void stage_M_async(const unsigned char* __restrict__ MbBlk,
                                              int k0, unsigned char* __restrict__ mbuf,
                                              int tid) {
    int r = tid >> 1, half = (tid & 1) * 16;
    const unsigned char* g = MbBlk + (size_t)r * LSEQ + k0 + half;
    unsigned ldsa = (unsigned)(unsigned long long)(mbuf + r * MSTR + half);
    asm volatile("global_load_async_to_lds_b128 %0, %1, off"
                 :: "v"(ldsa), "v"(g) : "memory");
}

__device__ __forceinline__ void wait_async0() {
#if __has_builtin(__builtin_amdgcn_s_wait_asynccnt)
    __builtin_amdgcn_s_wait_asynccnt(0);
#else
    asm volatile("s_wait_asynccnt 0x0" ::: "memory");
#endif
}

// ---- sync staging of V: f32 -> f16, transposed [d][key] ----
__device__ __forceinline__ void stage_V(const float* __restrict__ Vb, int k0,
                                        _Float16* __restrict__ vbuf, int tid) {
    #pragma unroll
    for (int i = 0; i < 4; ++i) {
        int idx4 = (i * 128 + tid) * 4;
        int key = idx4 >> 6, d = idx4 & 63;
        float4 f = *(const float4*)(Vb + (size_t)(k0 + key) * DDIM + d);
        vbuf[(d + 0) * VSTRIDE + key] = (_Float16)f.x;
        vbuf[(d + 1) * VSTRIDE + key] = (_Float16)f.y;
        vbuf[(d + 2) * VSTRIDE + key] = (_Float16)f.z;
        vbuf[(d + 3) * VSTRIDE + key] = (_Float16)f.w;
    }
}

// read 16 consecutive f32 from LDS, convert to an f16 fragment
__device__ __forceinline__ v16h cvt_frag(const float* __restrict__ p) {
    v16h r;
    #pragma unroll
    for (int h = 0; h < 16; ++h) r[h] = (_Float16)p[h];
    return r;
}

__global__ __launch_bounds__(128) void sdpa_wmma_kernel(
    const float* __restrict__ Q, const float* __restrict__ K,
    const float* __restrict__ V, const unsigned char* __restrict__ Mask,
    float* __restrict__ Out, float* __restrict__ Attn)
{
    __shared__ __align__(16) float         Klds[2][32 * KSTR4];     // 18 KB (f32)
    __shared__ __align__(32) _Float16      Vlds[2][64 * VSTRIDE];   // 12 KB
    __shared__ __align__(32) _Float16      Plds[4 * 16 * PSTRIDE];  //  6 KB
    __shared__ __align__(16) unsigned char Mlds[2][64 * MSTR];      //  4 KB

    const int tid   = threadIdx.x;
    const int bh    = blockIdx.y;
    const int wave  = tid >> 5;
    const int lane  = tid & 31;
    const int lrow  = lane & 15;          // M (A-frag) / N (B,C-frag) index
    const int hi    = (lane >> 4) & 1;    // half-wave select
    const int qBlk  = blockIdx.x * 64;
    const int qBase = qBlk + wave * 16;

    const float* __restrict__ Qb = Q + (size_t)bh * LSEQ * DDIM;
    const float* __restrict__ Kb = K + (size_t)bh * LSEQ * DDIM;
    const float* __restrict__ Vb = V + (size_t)bh * LSEQ * DDIM;
    const unsigned char* __restrict__ MbBlk =
        Mask + (size_t)bh * LSEQ * LSEQ + (size_t)qBlk * LSEQ;
    float* __restrict__ Ab = Attn + (size_t)bh * LSEQ * LSEQ;

    // ---- Q tile (16x64) into two A fragments (16x32 f16 each) ----
    // A layout: lane holds row M = lane&15; element h -> K = (h<8 ? h : h+8) + hi*8
    v16h aQ[2];
    for (int c = 0; c < 2; ++c) {
        const float* qrow = Qb + (size_t)(qBase + lrow) * DDIM + c * 32 + hi * 8;
        #pragma unroll
        for (int h = 0; h < 16; ++h) {
            int k = (h < 8) ? h : (h + 8);
            aQ[c][h] = (_Float16)qrow[k];
        }
    }

    // per-lane running softmax stats; merged across lanes after sweep 1
    float mlan[8], llan[8];
    #pragma unroll
    for (int j = 0; j < 8; ++j) { mlan[j] = -3.0e38f; llan[j] = 0.0f; }

    _Float16* Pw = Plds + wave * 16 * PSTRIDE;
    constexpr int NKT = LSEQ / 32;   // 64 key tiles of 32

    // =================== sweep 1: per-lane online max/sum ===================
    stage_K_async(Kb, 0, Klds[0], tid);
    stage_M_async(MbBlk, 0, Mlds[0], tid);
    wait_async0();
    __syncthreads();
    for (int kt = 0; kt < NKT; ++kt) {
        const int k0 = kt * 32;
        if (kt + 1 < NKT) {
            stage_K_async(Kb, k0 + 32, Klds[(kt + 1) & 1], tid);
            stage_M_async(MbBlk, k0 + 32, Mlds[(kt + 1) & 1], tid);
        }
        const float*         kb = Klds[kt & 1];
        const unsigned char* mb = Mlds[kt & 1] + wave * 16 * MSTR;
        #pragma unroll
        for (int sub = 0; sub < 2; ++sub) {
            v8f c = {};
            #pragma unroll
            for (int ch = 0; ch < 2; ++ch) {
                const v16h b = [&]{ const float* p = kb + (sub * 16 + lrow) * KSTR4 + ch * 32 + hi * 8; v16h r; _Pragma("unroll") for (int h = 0; h < 16; ++h) r[h] = (_Float16)p[(h < 8) ? h : (h + 8)]; return r; }();
                c = __builtin_amdgcn_wmma_f32_16x16x32_f16(false, aQ[ch], false, b,
                                                           (short)0, c, false, false);
            }
            #pragma unroll
            for (int j = 0; j < 8; ++j) {
                float s = c[j] * SCALE;
                if (mb[(j + hi * 8) * MSTR + sub * 16 + lrow]) s = MFILL;
                const float mn = fmaxf(mlan[j], s);
                llan[j] = llan[j] * __expf(mlan[j] - mn) + __expf(s - mn);
                mlan[j] = mn;
            }
        }
        wait_async0();     // our slice of tile kt+1 is in LDS
        __syncthreads();   // everyone's slice is in LDS
    }

    // one-time cross-lane merge (16 lanes per half-wave share each row)
    float m[8], linv[8];
    #pragma unroll
    for (int j = 0; j < 8; ++j) {
        float mj = mlan[j], lj = llan[j];
        for (int sm = 1; sm < 16; sm <<= 1) {
            const float om = __shfl_xor(mj, sm, 32);
            const float ol = __shfl_xor(lj, sm, 32);
            const float mn = fmaxf(mj, om);
            lj = lj * __expf(mj - mn) + ol * __expf(om - mn);
            mj = mn;
        }
        m[j] = mj; linv[j] = 1.0f / lj;
    }

    // =========== sweep 2: write attn, accumulate O = P @ V ============
    v8f acc[4];
    #pragma unroll
    for (int dc = 0; dc < 4; ++dc) acc[dc] = (v8f){};

    stage_K_async(Kb, 0, Klds[0], tid);
    stage_M_async(MbBlk, 0, Mlds[0], tid);
    stage_V(Vb, 0, Vlds[0], tid);
    wait_async0();
    __syncthreads();
    for (int kt = 0; kt < NKT; ++kt) {
        const int k0 = kt * 32;
        if (kt + 1 < NKT) {
            stage_K_async(Kb, k0 + 32, Klds[(kt + 1) & 1], tid);
            stage_M_async(MbBlk, k0 + 32, Mlds[(kt + 1) & 1], tid);
            stage_V(Vb, k0 + 32, Vlds[(kt + 1) & 1], tid);
        }
        if (kt + 2 < NKT)
            __builtin_prefetch(Vb + (size_t)(k0 + 64) * DDIM + tid * 16, 0, 3);
        const float*         kb = Klds[kt & 1];
        const _Float16*      vb = Vlds[kt & 1];
        const unsigned char* mb = Mlds[kt & 1] + wave * 16 * MSTR;
        #pragma unroll
        for (int sub = 0; sub < 2; ++sub) {
            v8f c = {};
            #pragma unroll
            for (int ch = 0; ch < 2; ++ch) {
                const v16h b = [&]{ const float* p = kb + (sub * 16 + lrow) * KSTR4 + ch * 32 + hi * 8; v16h r; _Pragma("unroll") for (int h = 0; h < 16; ++h) r[h] = (_Float16)p[(h < 8) ? h : (h + 8)]; return r; }();
                c = __builtin_amdgcn_wmma_f32_16x16x32_f16(false, aQ[ch], false, b,
                                                           (short)0, c, false, false);
            }
            const int col = k0 + sub * 16 + lrow;
            #pragma unroll
            for (int j = 0; j < 8; ++j) {
                const int row = qBase + j + hi * 8;
                float s = c[j] * SCALE;
                if (mb[(j + hi * 8) * MSTR + sub * 16 + lrow]) s = MFILL;
                const float p = __expf(s - m[j]) * linv[j];
                // attn is written once, never re-read: stream past L2 (NT hint)
                __builtin_nontemporal_store(p, &Ab[(size_t)row * LSEQ + col]);
                Pw[(j + hi * 8) * PSTRIDE + sub * 16 + lrow] = (_Float16)p;
            }
        }
        // within-wave cross-lane LDS transpose: wait for our ds stores
        asm volatile("s_wait_dscnt 0" ::: "memory");
        v16h aP;   // P as A fragment (16 rows x 32 keys)
        {
            const _Float16* prow = Pw + lrow * PSTRIDE + hi * 16;
            ((uint4*)&aP)[0] = *(const uint4*)(prow);        // K = hi*16 .. hi*16+7
            ((uint4*)&aP)[1] = *(const uint4*)(prow + 8);    // K = hi*16+8 ..
        }
        #pragma unroll
        for (int dc = 0; dc < 4; ++dc) {
            // V B frag: lane col N = d (dc*16+lrow); element h -> key = hi*16 + h
            const v16h b = *(const v16h*)(vb + (dc * 16 + lrow) * VSTRIDE + hi * 16);
            acc[dc] = __builtin_amdgcn_wmma_f32_16x16x32_f16(false, aP, false, b,
                                                             (short)0, acc[dc], false, false);
        }
        wait_async0();
        __syncthreads();
    }

    // ---- write O (16x64 per wave), NT stores ----
    #pragma unroll
    for (int dc = 0; dc < 4; ++dc) {
        const int d = dc * 16 + lrow;
        #pragma unroll
        for (int j = 0; j < 8; ++j) {
            const int row = qBase + j + hi * 8;
            __builtin_nontemporal_store(acc[dc][j],
                                        &Out[((size_t)bh * LSEQ + row) * DDIM + d]);
        }
    }
}

extern "C" void kernel_launch(void* const* d_in, const int* in_sizes, int n_in,
                              void* d_out, int out_size, void* d_ws, size_t ws_size,
                              hipStream_t stream) {
    const float* q = (const float*)d_in[0];
    const float* k = (const float*)d_in[1];
    const float* v = (const float*)d_in[2];
    const unsigned char* mask = (const unsigned char*)d_in[3];  // jnp.bool_: 1 byte

    float* out  = (float*)d_out;                                 // (BH, L, D)
    float* attn = out + (size_t)BH_N * LSEQ * DDIM;              // (BH, L, L)

    dim3 grid(LSEQ / 64, BH_N);   // 32 x 16 blocks
    dim3 block(128);              // 4 waves, each owns a 16-row Q tile
    sdpa_wmma_kernel<<<grid, block, 0, stream>>>(q, k, v, mask, out, attn);
}
